// DecodePredictions_56495999811781
// MI455X (gfx1250) — compile-verified, matrix operand order checked by reading, not run on hardware
//
#include <hip/hip_runtime.h>
#include <hip/hip_bf16.h>
#include <math.h>

#define BATCH 8
#define NANCH 76725
#define NCLS 80
#define KPRE 256
#define ROWSTRIDE 84   // 4 box regs + 80 class logits
#define CONF_THRV 0.05f
#define IOU_THRV 0.5f
#define MAX_PER_CLASS 100
#define MAX_DETN 100

typedef __attribute__((ext_vector_type(16))) _Float16 v16h;
typedef __attribute__((ext_vector_type(8)))  float    v8f;

__device__ __forceinline__ unsigned fkey(float f) {
  // order-preserving float -> uint key (larger float => larger key)
  unsigned u = __float_as_uint(f);
  return (u & 0x80000000u) ? ~u : (u | 0x80000000u);
}

__device__ __forceinline__ float sigmoidf(float x) { return 1.0f / (1.0f + expf(-x)); }

// descending bitonic sort of 256 (val, idx) pairs in LDS, 256 threads
__device__ void bitonic256_desc(float* v, int* id, int tid) {
  for (int ksz = 2; ksz <= 256; ksz <<= 1) {
    for (int j = ksz >> 1; j > 0; j >>= 1) {
      __syncthreads();
      int p = tid ^ j;
      if (p > tid) {
        float a = v[tid], b = v[p];
        bool descSeg = ((tid & ksz) == 0);
        bool sw = descSeg ? (a < b) : (a > b);
        if (sw) { v[tid] = b; v[p] = a; int t = id[tid]; id[tid] = id[p]; id[p] = t; }
      }
    }
  }
  __syncthreads();
}

// ---------------- Kernel 1: exact per-(b,c) top-256, 16-bit radix select ----------------
// Uses a 64K-bin histogram in the 320KB CDNA5 LDS: 2 histogram sweeps + 1
// collection sweep (vs 5 sweeps with 8-bit digits) over the strided logits.
__global__ __launch_bounds__(256) void topk_kernel(const float* __restrict__ pred,
                                                   int* __restrict__ cand_idx,
                                                   float* __restrict__ cand_score) {
  __shared__ unsigned hist16[65536];            // 256 KB (fits 320KB WGP LDS)
  __shared__ int bsum[256];
  __shared__ float sv[256]; __shared__ int si[256];
  __shared__ float ev[256]; __shared__ int ei[256];
  __shared__ int sh_target; __shared__ unsigned sh_prefix;
  __shared__ int cA, cE;
  const int tid = threadIdx.x;
  const int c = blockIdx.x, b = blockIdx.y;
  const float* base = pred + (size_t)b * NANCH * ROWSTRIDE + 4 + c;

  if (tid == 0) { sh_target = KPRE; sh_prefix = 0u; }

  for (int pass = 0; pass < 2; ++pass) {
    const int shift = 16 - pass * 16;           // 16 then 0
    // clear histogram
    for (int i = tid; i < 65536; i += 256) hist16[i] = 0u;
    __syncthreads();
    const unsigned pfx = sh_prefix;             // only bits >= shift+16 set
    for (int i = tid; i < NANCH; i += 256) {
      __builtin_prefetch(base + (size_t)(i + 2048) * ROWSTRIDE, 0, 1);  // global_prefetch_b8
      unsigned k = fkey(base[(size_t)i * ROWSTRIDE]);
      if (pass == 0) {
        atomicAdd(&hist16[k >> 16], 1u);
      } else if ((k & 0xFFFF0000u) == pfx) {
        atomicAdd(&hist16[k & 0xFFFFu], 1u);
      }
    }
    __syncthreads();
    // per-thread block sums (256 bins each)
    {
      int s = 0;
      const int o = tid * 256;
      for (int d = 0; d < 256; ++d) s += (int)hist16[o + d];
      bsum[tid] = s;
    }
    __syncthreads();
    if (tid == 0) {
      int cum = 0, tgt = sh_target;
      for (int bb = 255; bb >= 0; --bb) {
        cum += bsum[bb];
        if (cum >= tgt) {
          int tin = tgt - (cum - bsum[bb]);      // needed within block bb
          int cum2 = 0;
          const int o = bb * 256;
          for (int d = 255; d >= 0; --d) {
            int cc2 = (int)hist16[o + d];
            cum2 += cc2;
            if (cum2 >= tin) {
              sh_prefix |= ((unsigned)((bb << 8) | d)) << shift;
              sh_target = tin - (cum2 - cc2);    // >= 1 invariant
              break;
            }
          }
          break;
        }
      }
    }
    __syncthreads();
  }

  const unsigned T = sh_prefix;                 // exact key of the 256th largest
  if (tid == 0) { cA = 0; cE = 0; }
  __syncthreads();
  for (int i = tid; i < NANCH; i += 256) {
    float x = base[(size_t)i * ROWSTRIDE];
    unsigned k = fkey(x);
    if (k > T) { int p = atomicAdd(&cA, 1); sv[p] = x; si[p] = i; }          // p <= 254 guaranteed
    else if (k == T) { int q = atomicAdd(&cE, 1); if (q < KPRE) { ev[q] = x; ei[q] = i; } }
  }
  __syncthreads();
  const int nA = cA;
  if (tid >= nA) { int e = tid - nA; sv[tid] = ev[e]; si[tid] = ei[e]; }
  __syncthreads();
  sv[tid] = sigmoidf(sv[tid]);   // monotone: selection on logits == selection on scores
  __syncthreads();
  bitonic256_desc(sv, si, tid);
  size_t o = ((size_t)(b * NCLS + c)) * KPRE + tid;
  cand_score[o] = sv[tid];
  cand_idx[o] = si[tid];
}

// ---------------- Kernel 2: gather + box decode for selected candidates only ----------------
__global__ __launch_bounds__(256) void decode_kernel(const float* __restrict__ pred,
                                                     const float* __restrict__ anch,
                                                     const int* __restrict__ cand_idx,
                                                     float* __restrict__ cand_box) {
  size_t g = (size_t)blockIdx.x * 256 + threadIdx.x;  // over B*C*KPRE
  int bb = (int)(g / (NCLS * KPRE));
  int ai = cand_idx[g];
  const float* p = pred + ((size_t)bb * NANCH + (size_t)ai) * ROWSTRIDE;
  const float* an = anch + (size_t)ai * 4;
  float dx = p[0] * 0.1f, dy = p[1] * 0.1f, dw = p[2] * 0.2f, dh = p[3] * 0.2f;
  float ax = an[0], ay = an[1], aw = an[2], ah = an[3];
  float cx = dx * aw + ax, cy = dy * ah + ay;
  float bw = expf(dw) * aw, bh = expf(dh) * ah;
  cand_box[g * 4 + 0] = cx - bw * 0.5f;
  cand_box[g * 4 + 1] = cy - bh * 0.5f;
  cand_box[g * 4 + 2] = cx + bw * 0.5f;
  cand_box[g * 4 + 3] = cy + bh * 0.5f;
}

// ---------------- Kernel 3: async-LDS box staging + greedy NMS + WMMA prefix cap ----------------
__global__ __launch_bounds__(256) void nms_kernel(const float* __restrict__ cand_box,
                                                  const float* __restrict__ cand_score,
                                                  float* __restrict__ keep_out) {
  __shared__ float sbox[256 * 4];               // async-to-LDS staging (4 KB)
  __shared__ float sx1[256], sy1[256], sx2[256], sy2[256], sar[256], skp[256], spref[256];
  __shared__ float scs[16], soff[16];
  const int tid = threadIdx.x;
  const size_t base = (size_t)blockIdx.x * KPRE;

  // Stage this block's 256 boxes (16B/lane) straight into LDS via the CDNA5
  // async DMA path (ASYNCcnt), bypassing the VGPR round-trip.
  {
    const float* gptr = cand_box + (base + (size_t)tid) * 4;
    unsigned ldsoff = (unsigned)(size_t)(&sbox[tid * 4]);  // flat LDS ptr low32 == LDS byte addr
    asm volatile("global_load_async_to_lds_b128 %0, %1, off"
                 :: "v"(ldsoff), "v"(gptr) : "memory");
    asm volatile("s_wait_asynccnt 0" ::: "memory");
  }
  __syncthreads();

  float x1 = sbox[tid * 4 + 0];
  float y1 = sbox[tid * 4 + 1];
  float x2 = sbox[tid * 4 + 2];
  float y2 = sbox[tid * 4 + 3];
  sx1[tid] = x1; sy1[tid] = y1; sx2[tid] = x2; sy2[tid] = y2;
  sar[tid] = fmaxf(x2 - x1, 0.0f) * fmaxf(y2 - y1, 0.0f);
  skp[tid] = (cand_score[base + tid] > CONF_THRV) ? 1.0f : 0.0f;

  // greedy suppression: sequential over i (score-desc order), parallel over j>i
  for (int i = 0; i < KPRE - 1; ++i) {
    __syncthreads();
    if (skp[i] != 0.0f && tid > i) {
      float lx = fmaxf(sx1[i], sx1[tid]);
      float ly = fmaxf(sy1[i], sy1[tid]);
      float hx = fminf(sx2[i], sx2[tid]);
      float hy = fminf(sy2[i], sy2[tid]);
      float iw = fmaxf(hx - lx, 0.0f), ih = fmaxf(hy - ly, 0.0f);
      float inter = iw * ih;
      float iou = inter / (sar[i] + sar[tid] - inter + 1e-8f);
      if (iou > IOU_THRV) skp[tid] = 0.0f;
    }
  }
  __syncthreads();

  // Inclusive prefix-sum of keep flags via one v_wmma_f32_16x16x32_f16:
  //   D[m][n] = sum_{k<=m, k<16} keep[16*n + k]   (triangular A x tiled keep B)
  if (tid < 32) {  // wave 0, full EXEC
    const int lane = tid;
    const bool hi = lane >= 16;
    const int r = hi ? lane - 16 : lane;  // A row / B column / D column
    v16h a, bm;
    for (int h = 0; h < 16; ++h) {
      int v = h >> 1, s = h & 1;
      int k;
      if (v < 4) k = (hi ? 8 : 0) + 2 * v + s;
      else       k = (hi ? 24 : 16) + 2 * (v - 4) + s;
      a[h]  = (_Float16)((k <= r) ? 1.0f : 0.0f);                 // k>=16 -> 0 since r<=15
      bm[h] = (_Float16)((k < 16) ? skp[16 * r + k] : 0.0f);
    }
    v8f cc = {0.f, 0.f, 0.f, 0.f, 0.f, 0.f, 0.f, 0.f};
    v8f d = __builtin_amdgcn_wmma_f32_16x16x32_f16(false, a, false, bm,
                                                   (short)0, cc, false, false);
    for (int rr = 0; rr < 8; ++rr) {
      int m = hi ? rr + 8 : rr;           // D layout: lanes 0-15 -> M=rr, 16-31 -> M=rr+8
      spref[16 * r + m] = d[rr];          // within-chunk inclusive prefix
    }
  }
  __syncthreads();
  if (tid < 16) scs[tid] = spref[16 * tid + 15];  // chunk sums
  __syncthreads();
  if (tid == 0) { float run = 0.f; for (int n = 0; n < 16; ++n) { soff[n] = run; run += scs[n]; } }
  __syncthreads();
  float cum = spref[tid] + soff[tid >> 4];        // inclusive cumsum over 256
  float kf = (skp[tid] != 0.0f && cum <= (float)MAX_PER_CLASS) ? 1.0f : 0.0f;
  keep_out[base + tid] = kf;
}

// ---------------- Kernel 4: per-image global top-100 + output formatting ----------------
__global__ __launch_bounds__(256) void final_kernel(const float* __restrict__ cand_score,
                                                    const float* __restrict__ cand_box,
                                                    const float* __restrict__ keep,
                                                    float* __restrict__ out) {
  __shared__ unsigned hist[256];
  __shared__ float sv[256]; __shared__ int si[256];
  __shared__ float ev[256]; __shared__ int ei[256];
  __shared__ int sh_target; __shared__ unsigned sh_prefix;
  __shared__ int cA, cE, cnt;
  const int tid = threadIdx.x, b = blockIdx.x;
  const int N = NCLS * KPRE;
  const float* sc = cand_score + (size_t)b * N;
  const float* kp = keep + (size_t)b * N;
  const float* bx = cand_box + (size_t)b * N * 4;

  if (tid == 0) { sh_target = MAX_DETN; sh_prefix = 0u; cnt = 0; }
  for (int pass = 0; pass < 4; ++pass) {
    const int shift = 24 - pass * 8;
    const unsigned himask = (pass == 0) ? 0u : (0xFFFFFFFFu << (shift + 8));
    hist[tid] = 0u;
    __syncthreads();
    const unsigned pfx = sh_prefix;
    for (int i = tid; i < N; i += 256) {
      float v = (kp[i] > 0.5f) ? sc[i] : -1.0f;
      unsigned k = fkey(v);
      if ((k & himask) == pfx) atomicAdd(&hist[(k >> shift) & 0xFFu], 1u);
    }
    __syncthreads();
    if (tid == 0) {
      int cum = 0, tgt = sh_target;
      for (int d = 255; d >= 0; --d) {
        cum += (int)hist[d];
        if (cum >= tgt) {
          sh_prefix |= ((unsigned)d) << shift;
          sh_target = tgt - (cum - (int)hist[d]);
          break;
        }
      }
    }
    __syncthreads();
  }
  const unsigned T = sh_prefix;
  if (tid == 0) { cA = 0; cE = 0; }
  __syncthreads();
  for (int i = tid; i < N; i += 256) {
    float v = (kp[i] > 0.5f) ? sc[i] : -1.0f;
    unsigned k = fkey(v);
    if (k > T) { int p = atomicAdd(&cA, 1); if (p < MAX_DETN) { sv[p] = v; si[p] = i; } }
    else if (k == T) { int q = atomicAdd(&cE, 1); if (q < KPRE) { ev[q] = v; ei[q] = i; } }
  }
  __syncthreads();
  const int nA = cA;
  if (tid >= nA && tid < MAX_DETN) { int e = tid - nA; sv[tid] = ev[e]; si[tid] = ei[e]; }
  if (tid >= MAX_DETN) { sv[tid] = -1e30f; si[tid] = -1; }
  __syncthreads();
  bitonic256_desc(sv, si, tid);

  float* out_box = out;                                  // [B,100,4]
  float* out_sc  = out + BATCH * MAX_DETN * 4;           // [B,100]
  float* out_cls = out + BATCH * MAX_DETN * 5;           // [B,100]
  float* out_num = out + BATCH * MAX_DETN * 6;           // [B]
  if (tid < MAX_DETN) {
    float s = sv[tid];
    int fi = si[tid];                                    // flat index c*KPRE + k, always valid
    bool valid = s > 0.0f;
    float x1 = bx[(size_t)fi * 4 + 0], y1 = bx[(size_t)fi * 4 + 1];
    float x2 = bx[(size_t)fi * 4 + 2], y2 = bx[(size_t)fi * 4 + 3];
    float cx = (x1 + x2) * 0.5f, cy = (y1 + y2) * 0.5f;
    float w = x2 - x1, h = y2 - y1;
    size_t ob = ((size_t)b * MAX_DETN + tid) * 4;
    out_box[ob + 0] = valid ? cx : -1.0f;
    out_box[ob + 1] = valid ? cy : -1.0f;
    out_box[ob + 2] = valid ? w  : -1.0f;
    out_box[ob + 3] = valid ? h  : -1.0f;
    out_sc[b * MAX_DETN + tid]  = valid ? s : -1.0f;
    out_cls[b * MAX_DETN + tid] = valid ? (float)(fi / KPRE) : -1.0f;
    if (valid) atomicAdd(&cnt, 1);
  }
  __syncthreads();
  if (tid == 0) out_num[b] = (float)cnt;
}

extern "C" void kernel_launch(void* const* d_in, const int* in_sizes, int n_in,
                              void* d_out, int out_size, void* d_ws, size_t ws_size,
                              hipStream_t stream) {
  const float* pred = (const float*)d_in[0];   // [8, 76725, 84]
  const float* anch = (const float*)d_in[1];   // [76725, 4]
  float* out = (float*)d_out;

  char* ws = (char*)d_ws;
  const size_t nC = (size_t)BATCH * NCLS * KPRE;        // 163,840 candidates total
  int*   cand_idx   = (int*)ws;    ws += nC * sizeof(int);
  float* cand_score = (float*)ws;  ws += nC * sizeof(float);
  float* cand_box   = (float*)ws;  ws += nC * 4 * sizeof(float);
  float* keepbuf    = (float*)ws;  // nC floats; total workspace ~4.4 MB

  topk_kernel<<<dim3(NCLS, BATCH), 256, 0, stream>>>(pred, cand_idx, cand_score);
  decode_kernel<<<(int)(nC / 256), 256, 0, stream>>>(pred, anch, cand_idx, cand_box);
  nms_kernel<<<BATCH * NCLS, 256, 0, stream>>>(cand_box, cand_score, keepbuf);
  final_kernel<<<BATCH, 256, 0, stream>>>(cand_score, cand_box, keepbuf, out);
}
